// GCRNN_41772851921351
// MI455X (gfx1250) — matrix operand order repaired and checked by reading, not compiled, not run on hardware
//
#include <hip/hip_runtime.h>
#include <hip/hip_bf16.h>

typedef _Float16 h8   __attribute__((ext_vector_type(8)));
typedef _Float16 v16h __attribute__((ext_vector_type(16)));
typedef float    v8f  __attribute__((ext_vector_type(8)));

#define DIM    100
#define KPAD   256
#define GATES  400
#define TSTEPS 16
#define KCAND  9

__device__ __forceinline__ float sigf(float x) { return 1.0f / (1.0f + __expf(-x)); }

// ---------------- utility ----------------
__global__ void zero_f32(float* __restrict__ p, long n) {
    long i = (long)blockIdx.x * blockDim.x + threadIdx.x;
    if (i < n) p[i] = 0.0f;
}

// Build fused f16 weight matrix: Wcat[r, 0:100]=W_ih[r], [128:228]=W_hh[r], rest 0
__global__ void build_wcat(const float* __restrict__ Wih, const float* __restrict__ Whh,
                           _Float16* __restrict__ Wcat) {
    int i = blockIdx.x * blockDim.x + threadIdx.x;
    if (i >= GATES * KPAD) return;
    int r = i / KPAD, k = i % KPAD;
    float v = 0.0f;
    if (k < DIM)                      v = Wih[r * DIM + k];
    else if (k >= 128 && k < 128+DIM) v = Whh[r * DIM + (k - 128)];
    Wcat[i] = (_Float16)v;
}

// Gather h_prev (pre-aggregation) into xh[:,128:256] and c_prev (f32)
__global__ void gather_h(const float* __restrict__ emb, const float* __restrict__ c,
                         const int* __restrict__ seeds_t, _Float16* __restrict__ xh,
                         float* __restrict__ c_prev, int S) {
    int i = blockIdx.x * blockDim.x + threadIdx.x;
    if (i >= S * 128) return;
    int s = i >> 7, j = i & 127;
    int node = seeds_t[s];
    float v = (j < DIM) ? emb[(long)node * DIM + j] : 0.0f;
    xh[(long)s * KPAD + 128 + j] = (_Float16)v;
    if (j < DIM) c_prev[s * DIM + j] = c[(long)node * DIM + j];
}

// Gather x (post-aggregation) into xh[:,0:128]
__global__ void gather_x(const float* __restrict__ emb, const int* __restrict__ seeds_t,
                         _Float16* __restrict__ xh, int S) {
    int i = blockIdx.x * blockDim.x + threadIdx.x;
    if (i >= S * 128) return;
    int s = i >> 7, j = i & 127;
    int node = seeds_t[s];
    float v = (j < DIM) ? emb[(long)node * DIM + j] : 0.0f;
    xh[(long)s * KPAD + j] = (_Float16)v;
}

// One wave per edge: msg = emb[src]*rel[cat], atomic-add into agg[dst], count dst
__global__ void edge_kernel(const float* __restrict__ emb, const float* __restrict__ rel,
                            const int* __restrict__ src, const int* __restrict__ dst,
                            const int* __restrict__ cat, float* __restrict__ agg,
                            float* __restrict__ cnt, int E) {
    long gid = (long)blockIdx.x * blockDim.x + threadIdx.x;
    int e = (int)(gid >> 5);
    int l = (int)(gid & 31);
    if (e >= E) return;
    int s  = src[e];
    int d  = dst[e];
    int ct = cat[e];
    const float* es = emb + (long)s  * DIM;
    const float* rc = rel + (long)ct * DIM;
    float*       ad = agg + (long)d  * DIM;
    #pragma unroll
    for (int j = l; j < DIM; j += 32)
        atomicAdd(&ad[j], es[j] * rc[j]);
    if (l == 0) atomicAdd(&cnt[d], 1.0f);
}

// emb += agg / max(cnt,1); zero agg in the same pass
__global__ void update_emb(float* __restrict__ emb, float* __restrict__ agg,
                           const float* __restrict__ cnt, long ND) {
    long i = (long)blockIdx.x * blockDim.x + threadIdx.x;
    if (i >= ND) return;
    float a = agg[i];
    agg[i] = 0.0f;
    float cn = cnt[i / DIM];
    emb[i] += a / fmaxf(cn, 1.0f);
}

// gates[S,400] = [x|h] (f16, K=256) @ Wcat^T, f32 accumulate via WMMA.
// One wave per 16x16 output tile; 8 chained v_wmma_f32_16x16x32_f16 per wave.
__global__ void lstm_gemm(const _Float16* __restrict__ xh, const _Float16* __restrict__ Wcat,
                          float* __restrict__ gates, int S) {
    int wave = (int)(((long)blockIdx.x * blockDim.x + threadIdx.x) >> 5);
    int lane = threadIdx.x & 31;
    const int nTilesN = GATES / 16;             // 25
    int totalTiles = (S / 16) * nTilesN;
    if (wave >= totalTiles) return;
    int mt = wave / nTilesN;
    int nt = wave % nTilesN;

    // A fragment source: lane L and L+16 both hold row (mt*16 + L&15);
    // lanes 0-15 cover K {0..15}+k0, lanes 16-31 cover K {16..31}+k0.
    int arow = mt * 16 + (lane & 15);
    const _Float16* abase = xh + (long)arow * KPAD + ((lane >> 4) & 1) * 16;

    // B fragment: B = Wcat^T. Lane holds column n=lane&15 of the tile, i.e. row
    // (nt*16 + n) of Wcat; 16 contiguous K per lane (lo/hi half by lane group).
    int brow = nt * 16 + (lane & 15);
    const _Float16* bbase = Wcat + (long)brow * KPAD + ((lane >> 4) & 1) * 16;

    v8f acc = {};
    #pragma unroll
    for (int k0 = 0; k0 < KPAD; k0 += 32) {
        h8 lo = *(const h8*)(abase + k0);        // K = kbase .. kbase+7
        h8 hi = *(const h8*)(abase + k0 + 8);    // K = kbase+8 .. kbase+15
        v16h a = __builtin_shufflevector(lo, hi, 0,1,2,3,4,5,6,7,8,9,10,11,12,13,14,15);
        v16h b = *(const v16h*)(bbase + k0);
        acc = __builtin_amdgcn_wmma_f32_16x16x32_f16(
            /*neg_a=*/false, a, /*neg_b=*/false, b,
            /*c_mod=*/(short)0, acc, /*reuse_a=*/false, /*reuse_b=*/false);
    }

    // C/D layout: VGPR r -> M = r (lanes 0-15) or 8+r (lanes 16-31); N = lane&15
    int rowbase = mt * 16 + ((lane >> 4) & 1) * 8;
    int coln    = nt * 16 + (lane & 15);
    #pragma unroll
    for (int r = 0; r < 8; ++r)
        gates[(long)(rowbase + r) * GATES + coln] = acc[r];
}

// LSTM nonlinearity + scatter back to emb/c + record user embedding
__global__ void lstm_elem(const float* __restrict__ gates, const float* __restrict__ bih,
                          const float* __restrict__ bhh, const float* __restrict__ c_prev,
                          const int* __restrict__ seeds_t, float* __restrict__ emb,
                          float* __restrict__ c, float* __restrict__ user, int S, int t) {
    int i = blockIdx.x * blockDim.x + threadIdx.x;
    if (i >= S * DIM) return;
    int s = i / DIM, d = i % DIM;
    const float* g = gates + (long)s * GATES;
    float ig = g[d]           + bih[d]           + bhh[d];
    float fg = g[DIM + d]     + bih[DIM + d]     + bhh[DIM + d];
    float gg = g[2*DIM + d]   + bih[2*DIM + d]   + bhh[2*DIM + d];
    float og = g[3*DIM + d]   + bih[3*DIM + d]   + bhh[3*DIM + d];
    float cp = c_prev[i];
    float cn = sigf(fg) * cp + sigf(ig) * tanhf(gg);
    float hn = sigf(og) * tanhf(cn);
    int node = seeds_t[s];
    emb[(long)node * DIM + d] = hn;
    c[(long)node * DIM + d]   = cn;
    user[((long)t * S + s) * DIM + d] = hn;
}

// One wave per user row: 9 gathered dot-products, shfl reduction
__global__ void score_kernel(const float* __restrict__ user, const float* __restrict__ emb,
                             const int* __restrict__ cand, float* __restrict__ out, long NU) {
    long wave = ((long)blockIdx.x * blockDim.x + threadIdx.x) >> 5;
    int lane = threadIdx.x & 31;
    if (wave >= NU) return;
    const float* u = user + wave * DIM;
    float u0 = u[lane], u1 = u[lane + 32], u2 = u[lane + 64];
    float u3 = (lane < 4) ? u[lane + 96] : 0.0f;
    for (int k = 0; k < KCAND; ++k) {
        int ci = cand[wave * KCAND + k];
        const float* cr = emb + (long)ci * DIM;
        float p = u0 * cr[lane] + u1 * cr[lane + 32] + u2 * cr[lane + 64];
        if (lane < 4) p += u3 * cr[lane + 96];
        #pragma unroll
        for (int m = 16; m >= 1; m >>= 1) p += __shfl_xor(p, m, 32);
        if (lane == 0) out[wave * KCAND + k] = p;
    }
}

extern "C" void kernel_launch(void* const* d_in, const int* in_sizes, int n_in,
                              void* d_out, int out_size, void* d_ws, size_t ws_size,
                              hipStream_t stream) {
    const float* node_emb = (const float*)d_in[0];
    const float* cx       = (const float*)d_in[1];
    const float* rel      = (const float*)d_in[2];
    const float* Wih      = (const float*)d_in[3];
    const float* Whh      = (const float*)d_in[4];
    const float* bih      = (const float*)d_in[5];
    const float* bhh      = (const float*)d_in[6];
    const int*   src      = (const int*)d_in[7];
    const int*   dst      = (const int*)d_in[8];
    const int*   cat      = (const int*)d_in[9];
    const int*   seeds    = (const int*)d_in[10];
    const int*   cand     = (const int*)d_in[11];

    const int  D = DIM;
    const long N = in_sizes[0] / D;
    const int  T = TSTEPS;
    const long E = in_sizes[7] / T;
    const int  S = in_sizes[10] / T;
    const long ND = N * D;

    // Carve workspace (256B-aligned regions)
    char* w = (char*)d_ws;
    auto carve = [&](size_t bytes) -> char* {
        char* p = w;
        w += (bytes + 255) & ~(size_t)255;
        return p;
    };
    float*    emb   = (float*)carve((size_t)ND * 4);
    float*    cbuf  = (float*)carve((size_t)ND * 4);
    float*    agg   = (float*)carve((size_t)ND * 4);
    float*    cnt   = (float*)carve((size_t)N * 4);
    float*    cprev = (float*)carve((size_t)S * D * 4);
    _Float16* xh    = (_Float16*)carve((size_t)S * KPAD * 2);
    _Float16* Wc    = (_Float16*)carve((size_t)GATES * KPAD * 2);
    float*    gates = (float*)carve((size_t)S * GATES * 4);
    float*    user  = (float*)carve((size_t)T * S * D * 4);
    (void)ws_size; (void)n_in; (void)out_size;

    // Initialize mutable state (do not touch inputs)
    hipMemcpyAsync(emb,  node_emb, (size_t)ND * 4, hipMemcpyDeviceToDevice, stream);
    hipMemcpyAsync(cbuf, cx,       (size_t)ND * 4, hipMemcpyDeviceToDevice, stream);
    zero_f32<<<(unsigned)((ND + 255) / 256), 256, 0, stream>>>(agg, ND);
    zero_f32<<<(unsigned)((N + 255) / 256), 256, 0, stream>>>(cnt, N);
    build_wcat<<<(GATES * KPAD + 255) / 256, 256, 0, stream>>>(Wih, Whh, Wc);

    const int gemmWaves = (S / 16) * (GATES / 16);
    for (int t = 0; t < T; ++t) {
        const int* seeds_t = seeds + (long)t * S;
        gather_h<<<(S * 128 + 255) / 256, 256, 0, stream>>>(emb, cbuf, seeds_t, xh, cprev, S);
        edge_kernel<<<(unsigned)((E * 32 + 255) / 256), 256, 0, stream>>>(
            emb, rel, src + t * E, dst + t * E, cat + t * E, agg, cnt, (int)E);
        update_emb<<<(unsigned)((ND + 255) / 256), 256, 0, stream>>>(emb, agg, cnt, ND);
        zero_f32<<<(unsigned)((N + 255) / 256), 256, 0, stream>>>(cnt, N);
        gather_x<<<(S * 128 + 255) / 256, 256, 0, stream>>>(emb, seeds_t, xh, S);
        lstm_gemm<<<(gemmWaves * 32 + 255) / 256, 256, 0, stream>>>(xh, Wc, gates, S);
        lstm_elem<<<(S * D + 255) / 256, 256, 0, stream>>>(
            gates, bih, bhh, cprev, seeds_t, emb, cbuf, user, S, t);
    }

    const long NU = (long)T * S;
    score_kernel<<<(unsigned)((NU * 32 + 255) / 256), 256, 0, stream>>>(
        user, emb, cand, (float*)d_out, NU);
}
